// Network_63763084476816
// MI455X (gfx1250) — compile-verified, matrix operand order checked by reading, not run on hardware
//
#include <hip/hip_runtime.h>
#include <hip/hip_bf16.h>

// ---------------------------------------------------------------------------
// MI455X (gfx1250) implementation. wave32, WMMA bf16 (f32 accum).
//
// Pipeline:
//   prep      : analytic graph aggregation (topology is fixed) -> x_bf16, agg_bf16
//   tconv x3  : f32 -> bf16 transposed weights (N-major)
//   gemm_gnn  : h = relu(x@W_self + agg@W_msg + b_g)
//               weights staged to LDS once per block via async-to-LDS;
//               per k-step: batch-load 2 A + 8 B fragments, then 8 WMMAs
//   pool      : combined = [h_clin | mean(h_pix)]  (bf16)
//   gemm_mlp  : z = relu(combined@W1 + b1); per k-step 1 A + 4 B, 4 WMMAs
//   final     : out = z@W2 + b2   (wave reduction)
// ---------------------------------------------------------------------------

#define N_CLIN  38
#define N_PIX   36
#define N_NODES 74
#define FV      128
#define BATCH   1024
#define HID     512
#define M_GNN   (BATCH * N_NODES)      // 75776 rows
#define K_MLP   (FV * (N_CLIN + 1))    // 4992 = 156 * 32

typedef __attribute__((ext_vector_type(16))) __bf16 v16bf;
typedef __attribute__((ext_vector_type(8)))  float  v8f;

union FragU { uint4 u[2]; v16bf v; };

__device__ __forceinline__ v8f wmma_bf16(v16bf a, v16bf b, v8f c) {
    return __builtin_amdgcn_wmma_f32_16x16x32_bf16(false, a, false, b,
                                                   (short)0, c, false, false);
}

// Load one 16x32 bf16 fragment (A or B role) from a row-major bf16 matrix
// (global or LDS; address space inferred after inlining).
// Matches the CDNA5 16-bit A-matrix layout (ISA 7.12.2):
//   lane<16 : row = row0+lane,    K = k0+0..7  (v0..3), k0+16..23 (v4..7)
//   lane>=16: row = row0+lane-16, K = k0+8..15 (v0..3), k0+24..31 (v4..7)
__device__ __forceinline__ v16bf load_frag16(const __hip_bfloat16* __restrict__ base,
                                             int row0, int ld, int k0, int lane) {
    int r  = row0 + (lane & 15);
    int ko = k0 + ((lane & 16) ? 8 : 0);
    const __hip_bfloat16* p = base + (size_t)r * ld + ko;
    FragU f;
    f.u[0] = *(const uint4*)(p);        // 8 bf16 : K = ko .. ko+7
    f.u[1] = *(const uint4*)(p + 16);   // 8 bf16 : K = ko+16 .. ko+23
    return f.v;
}

// CDNA5 async copy: one 16-byte chunk, global -> LDS (per-lane addresses).
// Tracked with ASYNCcnt (CDNA5 ISA).
__device__ __forceinline__ void async_g2lds_b128(unsigned lds_addr, const void* gaddr) {
    asm volatile("global_load_async_to_lds_b128 %0, %1, off"
                 :: "v"(lds_addr), "v"(gaddr) : "memory");
}
__device__ __forceinline__ void wait_asynccnt0() {
    asm volatile("s_wait_asynccnt 0" ::: "memory");
}

// --------------------------- prep: graph aggregation ------------------------
__global__ void prep_kernel(const float* __restrict__ clin,
                            const float* __restrict__ img,
                            __hip_bfloat16* __restrict__ xb,
                            __hip_bfloat16* __restrict__ aggb) {
    __shared__ float xs[N_NODES * FV];     // 37.9 KB of the 320 KB LDS
    const int b = blockIdx.x;              // one graph per block
    const int f = threadIdx.x;             // 128 threads = feature lanes
    const float* cp = clin + (size_t)b * N_CLIN * FV + f;
    const float* ip = img  + (size_t)b * N_PIX  * FV + f;
    float sc = 0.f, sp = 0.f;
    #pragma unroll 4
    for (int r = 0; r < N_CLIN; ++r) { float v = cp[(size_t)r * FV]; xs[r * FV + f] = v; sc += v; }
    #pragma unroll 4
    for (int r = 0; r < N_PIX;  ++r) { float v = ip[(size_t)r * FV]; xs[(N_CLIN + r) * FV + f] = v; sp += v; }
    // each thread only touches its own column f -> no barrier needed
    __hip_bfloat16* xo = xb   + (size_t)b * N_NODES * FV + f;
    __hip_bfloat16* ao = aggb + (size_t)b * N_NODES * FV + f;
    const float inv37 = 1.f / 37.f, inv39 = 1.f / 39.f;
    #pragma unroll 2
    for (int r = 0; r < N_NODES; ++r) {
        float v = xs[r * FV + f];
        xo[(size_t)r * FV] = __float2bfloat16(v);
        float a = (r < N_CLIN) ? (v + sp) * inv37 : (v + sc) * inv39;
        ao[(size_t)r * FV] = __float2bfloat16(a);
    }
}

// ------------------- transpose + convert weights to bf16 --------------------
// dst[n*K + k] = bf16(src[k*N + n])
__global__ void tconv_kernel(const float* __restrict__ src,
                             __hip_bfloat16* __restrict__ dst, int K, int N) {
    int total = K * N;
    for (int idx = blockIdx.x * blockDim.x + threadIdx.x; idx < total;
         idx += gridDim.x * blockDim.x) {
        int n = idx / K, k = idx - n * K;
        dst[idx] = __float2bfloat16(src[(size_t)k * N + n]);
    }
}

// ----------------------- GNN layer: fused double GEMM -----------------------
// h[M_GNN x 128] = relu(x@W_self + agg@W_msg + b_g), bf16 output.
// Block: 8 waves = 4 M-tiles x 2 N-halves (64Mx128N). Weights live in LDS.
__global__ void gemm_gnn_kernel(const __hip_bfloat16* __restrict__ xb,
                                const __hip_bfloat16* __restrict__ aggb,
                                const __hip_bfloat16* __restrict__ wts,  // 128x128 N-major
                                const __hip_bfloat16* __restrict__ wtm,  // 128x128 N-major
                                const float* __restrict__ bg,
                                __hip_bfloat16* __restrict__ h) {
    __shared__ __align__(16) __hip_bfloat16 smem[2 * FV * FV];   // 64 KB
    const int tid = threadIdx.x;

    // Stage both weight matrices into LDS with CDNA5 async copies:
    // 2 x 32KB = 4096 x b128, 256 threads -> 16 async ops per lane.
    {
        const uint4* g0 = (const uint4*)wts;
        const uint4* g1 = (const uint4*)wtm;
        // Within the LDS aperture, the low 32 bits of the generic address
        // are the LDS byte offset (ISA 10.2 aperture mapping).
        unsigned base = (unsigned)(uintptr_t)(&smem[0]);
        #pragma unroll
        for (int i = 0; i < 8; ++i) {
            int idx = tid + i * 256;                       // 0..2047
            async_g2lds_b128(base + idx * 16,           g0 + idx);
            async_g2lds_b128(base + 32768 + idx * 16,   g1 + idx);
        }
        wait_asynccnt0();
    }
    __syncthreads();

    const int lane = tid & 31;
    const int wave = tid >> 5;
    const int m0   = blockIdx.x * 64 + (wave & 3) * 16;  // 1184 blocks
    const int n0   = (wave >> 2) * 64;                   // N strip: 0 or 64
    const __hip_bfloat16* lws = smem;                    // W_self^T in LDS
    const __hip_bfloat16* lwm = smem + FV * FV;          // W_msg^T in LDS

    v8f acc[4] = {};
    #pragma unroll
    for (int kk = 0; kk < FV; kk += 32) {
        // Batch-load every fragment of this k-step into distinct registers
        // so the ds/global loads clause together under one wait, then the
        // 8 WMMAs issue back-to-back into the XDL pipe.
        v16bf ax = load_frag16(xb,   m0, FV, kk, lane);
        v16bf aa = load_frag16(aggb, m0, FV, kk, lane);
        v16bf bs[4], bm[4];
        #pragma unroll
        for (int j = 0; j < 4; ++j) {
            bs[j] = load_frag16(lws, n0 + j * 16, FV, kk, lane);
            bm[j] = load_frag16(lwm, n0 + j * 16, FV, kk, lane);
        }
        #pragma unroll
        for (int j = 0; j < 4; ++j) acc[j] = wmma_bf16(ax, bs[j], acc[j]);
        #pragma unroll
        for (int j = 0; j < 4; ++j) acc[j] = wmma_bf16(aa, bm[j], acc[j]);
    }

    // C/D layout: VGPR r, lane -> (M = r + 8*(lane>=16), N = lane&15)
    const int nb = n0 + (lane & 15);
    const int mb = m0 + ((lane & 16) ? 8 : 0);
    #pragma unroll
    for (int j = 0; j < 4; ++j) {
        const int n = nb + j * 16;
        const float bias = bg[n];
        #pragma unroll
        for (int r = 0; r < 8; ++r) {
            float v = acc[j][r] + bias;
            v = v > 0.f ? v : 0.f;
            h[(size_t)(mb + r) * FV + n] = __float2bfloat16(v);
        }
    }
}

// ----------------------- pooling: build MLP input ---------------------------
__global__ void pool_kernel(const __hip_bfloat16* __restrict__ h,
                            __hip_bfloat16* __restrict__ comb) {
    const int b = blockIdx.x;
    const int f = threadIdx.x;  // 128
    const __hip_bfloat16* hb = h + (size_t)b * N_NODES * FV;
    __hip_bfloat16* cb = comb + (size_t)b * K_MLP;
    #pragma unroll 4
    for (int r = 0; r < N_CLIN; ++r) cb[r * FV + f] = hb[(size_t)r * FV + f];
    float s = 0.f;
    #pragma unroll 4
    for (int r = N_CLIN; r < N_NODES; ++r) s += __bfloat162float(hb[(size_t)r * FV + f]);
    cb[N_CLIN * FV + f] = __float2bfloat16(s * (1.f / (float)N_PIX));
}

// --------------------------- MLP layer 1 GEMM -------------------------------
// z[1024 x 512] = relu(combined @ W1 + b1), f32 output.
// Block: 8 waves share one 16-row A tile and cover the full N=512
// (one 64-wide strip per wave, 4 accumulators).
__global__ void gemm_mlp_kernel(const __hip_bfloat16* __restrict__ comb, // 1024 x 4992
                                const __hip_bfloat16* __restrict__ wt1,  // 512  x 4992 (N-major)
                                const float* __restrict__ b1,
                                float* __restrict__ z) {
    const int lane = threadIdx.x & 31;
    const int wave = threadIdx.x >> 5;
    const int m0   = blockIdx.x * 16;   // 64 blocks = 64 M-tiles
    const int n0   = wave * 64;         // 8 waves cover N = 512
    v8f acc[4] = {};
    #pragma unroll 2
    for (int kk = 0; kk < K_MLP; kk += 32) {
        if (kk + 32 < K_MLP) {  // pull next k-tile toward the caches
            __builtin_prefetch(comb + (size_t)(m0 + (lane & 15)) * K_MLP + kk + 32, 0, 0);
            __builtin_prefetch(wt1  + (size_t)(n0 + (lane & 15)) * K_MLP + kk + 32, 0, 0);
        }
        // Batch all loads of the k-step, then burst the WMMAs.
        v16bf a = load_frag16(comb, m0, K_MLP, kk, lane);
        v16bf b[4];
        #pragma unroll
        for (int j = 0; j < 4; ++j) b[j] = load_frag16(wt1, n0 + j * 16, K_MLP, kk, lane);
        #pragma unroll
        for (int j = 0; j < 4; ++j) acc[j] = wmma_bf16(a, b[j], acc[j]);
    }
    const int nb = n0 + (lane & 15);
    const int mb = m0 + ((lane & 16) ? 8 : 0);
    #pragma unroll
    for (int j = 0; j < 4; ++j) {
        const int n = nb + j * 16;
        const float bias = b1[n];
        #pragma unroll
        for (int r = 0; r < 8; ++r) {
            float v = acc[j][r] + bias;
            z[(size_t)(mb + r) * HID + n] = v > 0.f ? v : 0.f;
        }
    }
}

// --------------------------- final 512 -> 1 dot -----------------------------
__global__ void final_kernel(const float* __restrict__ z,
                             const float* __restrict__ w2,
                             const float* __restrict__ b2,
                             float* __restrict__ out) {
    const int lane = threadIdx.x & 31;
    const int wave = threadIdx.x >> 5;
    const int b    = blockIdx.x * 8 + wave;   // one sample per wave
    const float* zr = z + (size_t)b * HID;
    float acc = 0.f;
    #pragma unroll
    for (int i = 0; i < HID / 32; ++i) acc += zr[lane + i * 32] * w2[lane + i * 32];
    #pragma unroll
    for (int off = 16; off > 0; off >>= 1) acc += __shfl_down(acc, off, 32);
    if (lane == 0) out[b] = acc + b2[0];
}

// ---------------------------------------------------------------------------
extern "C" void kernel_launch(void* const* d_in, const int* in_sizes, int n_in,
                              void* d_out, int out_size, void* d_ws, size_t ws_size,
                              hipStream_t stream) {
    (void)in_sizes; (void)n_in; (void)out_size; (void)ws_size;
    const float* clin  = (const float*)d_in[0];  // (B,38,128)
    const float* img   = (const float*)d_in[1];  // (B,36,128)
    const float* Wself = (const float*)d_in[2];  // (128,128)
    const float* Wmsg  = (const float*)d_in[3];  // (128,128)
    const float* bg    = (const float*)d_in[4];  // (128)
    const float* W1    = (const float*)d_in[5];  // (4992,512)
    const float* b1    = (const float*)d_in[6];  // (512)
    const float* W2    = (const float*)d_in[7];  // (512,1)
    const float* b2    = (const float*)d_in[8];  // (1)
    // d_in[9] = edge_index: topology is fixed/analytic -> unused
    float* out = (float*)d_out;

    char*  ws  = (char*)d_ws;
    size_t off = 0;
    auto alloc = [&](size_t bytes) -> void* {
        void* p = ws + off;
        off = (off + bytes + 255) & ~(size_t)255;
        return p;
    };
    __hip_bfloat16* xb   = (__hip_bfloat16*)alloc((size_t)M_GNN * FV * 2);   // 19.4 MB
    __hip_bfloat16* aggb = (__hip_bfloat16*)alloc((size_t)M_GNN * FV * 2);   // 19.4 MB
    __hip_bfloat16* wts  = (__hip_bfloat16*)alloc((size_t)FV * FV * 2);
    __hip_bfloat16* wtm  = (__hip_bfloat16*)alloc((size_t)FV * FV * 2);
    __hip_bfloat16* wt1  = (__hip_bfloat16*)alloc((size_t)HID * K_MLP * 2);  // 5.1 MB
    __hip_bfloat16* hbuf = (__hip_bfloat16*)alloc((size_t)M_GNN * FV * 2);   // 19.4 MB
    __hip_bfloat16* comb = (__hip_bfloat16*)alloc((size_t)BATCH * K_MLP * 2);// 10.2 MB
    float*          zbuf = (float*)alloc((size_t)BATCH * HID * 4);           // 2.0 MB

    // 1) graph aggregation + bf16 conversion
    prep_kernel<<<BATCH, FV, 0, stream>>>(clin, img, xb, aggb);

    // 2) weight transpose-converts
    tconv_kernel<<<64,   256, 0, stream>>>(Wself, wts, FV, FV);
    tconv_kernel<<<64,   256, 0, stream>>>(Wmsg,  wtm, FV, FV);
    tconv_kernel<<<1024, 256, 0, stream>>>(W1,    wt1, K_MLP, HID);

    // 3) GNN fused GEMM: 1184 blocks x 8 waves, 64Mx128N per block
    gemm_gnn_kernel<<<M_GNN / 64, 256, 0, stream>>>(xb, aggb, wts, wtm, bg, hbuf);

    // 4) pooling
    pool_kernel<<<BATCH, FV, 0, stream>>>(hbuf, comb);

    // 5) MLP layer 1 GEMM: 64 blocks, each 16M x 512N
    gemm_mlp_kernel<<<64, 256, 0, stream>>>(comb, wt1, b1, zbuf);

    // 6) final dot product
    final_kernel<<<BATCH / 8, 256, 0, stream>>>(zbuf, W2, b2, out);
}
